// LSAMCell_5076651344123
// MI455X (gfx1250) — matrix operand order repaired and checked by reading, not compiled, not run on hardware
//
#include <hip/hip_runtime.h>
#include <hip/hip_bf16.h>
#include <math.h>

typedef __attribute__((ext_vector_type(16))) __bf16 v16bf;
typedef __attribute__((ext_vector_type(8)))  __bf16 v8bf;
typedef __attribute__((ext_vector_type(8)))  float  v8f;
typedef int v4i __attribute__((vector_size(4 * sizeof(int))));

union ABFrag { v16bf v; v8bf h[2]; };

static __device__ __forceinline__ unsigned short f32_to_bf16_rne(float f) {
    unsigned int u = __float_as_uint(f);
    unsigned int r = u + 0x7FFFu + ((u >> 16) & 1u);   // round-to-nearest-even
    return (unsigned short)(r >> 16);
}

// --- gfx1250 async global->LDS copy (ASYNCcnt path), with safe fallback ----
static __device__ __forceinline__ void async_copy16(void* lds, const void* g) {
#if __has_builtin(__builtin_amdgcn_global_load_async_to_lds_b128)
    __builtin_amdgcn_global_load_async_to_lds_b128(
        (__attribute__((address_space(1))) v4i*)(v4i*)(void*)g,
        (__attribute__((address_space(3))) v4i*)(v4i*)lds,
        0, 0);
#else
    *(uint4*)lds = *(const uint4*)g;
#endif
}
static __device__ __forceinline__ void wait_async_all() {
#if __has_builtin(__builtin_amdgcn_global_load_async_to_lds_b128)
 #if __has_builtin(__builtin_amdgcn_s_wait_asynccnt)
    __builtin_amdgcn_s_wait_asynccnt(0);
 #else
    asm volatile("s_wait_asynccnt 0" ::: "memory");
 #endif
#endif
}

// ---------------------------------------------------------------------------
// Kernel 0a: xh_bf16[b][k] = bf16(concat(x,h))   (1024 x 2048), packed stores
// ---------------------------------------------------------------------------
__global__ __launch_bounds__(256) void pack_xh_bf16(
    const float* __restrict__ x, const float* __restrict__ h,
    unsigned short* __restrict__ xh)
{
    int idx = blockIdx.x * 256 + threadIdx.x;      // one bf16-pair per thread
    int row = idx >> 10;                           // 1024 pairs per row
    int col = (idx & 1023) * 2;
    float a, b;
    if (col < 1024) { a = x[(size_t)row * 1024 + col];        b = x[(size_t)row * 1024 + col + 1]; }
    else            { a = h[(size_t)row * 1024 + col - 1024]; b = h[(size_t)row * 1024 + col - 1023]; }
    unsigned int packed = (unsigned int)f32_to_bf16_rne(a)
                        | ((unsigned int)f32_to_bf16_rne(b) << 16);
    ((unsigned int*)xh)[idx] = packed;
}

// ---------------------------------------------------------------------------
// Kernel 0b: Wt_bf16[n][k] = bf16(Wqkv[k][n])  (3072 x 2048), LDS-tiled
// transpose so the GEMM's B tile is K-contiguous (matches B fragment layout).
// ---------------------------------------------------------------------------
__global__ __launch_bounds__(256) void transpose_w_bf16(
    const float* __restrict__ W, unsigned short* __restrict__ Wt)
{
    __shared__ unsigned short tile[32][33];        // pad to dodge bank conflicts
    const int tk = blockIdx.y * 32;                // k-block (rows of W)
    const int tn = blockIdx.x * 32;                // n-block (cols of W)
    const int c = threadIdx.x & 31, r8 = threadIdx.x >> 5;
#pragma unroll
    for (int i = 0; i < 4; ++i) {
        int r = r8 + i * 8;
        tile[r][c] = f32_to_bf16_rne(W[(size_t)(tk + r) * 3072 + tn + c]);
    }
    __syncthreads();
#pragma unroll
    for (int i = 0; i < 4; ++i) {
        int r = r8 + i * 8;
        Wt[(size_t)(tn + r) * 2048 + tk + c] = tile[c][r];
    }
}

// ---------------------------------------------------------------------------
// Kernel 1: qkv = xh @ Wqkv + b  via bf16 WMMA, async-to-LDS double buffering.
// Block tile 128x128, BK=32, 8 waves; wave tile 64x32 = 4x2 WMMA accumulators.
// ---------------------------------------------------------------------------
#define BM 128
#define BN 128
#define BK 32

__global__ __launch_bounds__(256) void qkv_gemm(
    const unsigned short* __restrict__ xh_bf,   // (1024, 2048) bf16, [m][k]
    const unsigned short* __restrict__ wt_bf,   // (3072, 2048) bf16, [n][k]
    const float* __restrict__ bias,
    float* __restrict__ out)
{
    __shared__ __align__(16) unsigned short As[2][BM][BK];   // 2 x 8KB
    __shared__ __align__(16) unsigned short Bs[2][BN][BK];   // 2 x 8KB

    const int tid  = threadIdx.x;
    const int bm   = blockIdx.y * BM;
    const int bn   = blockIdx.x * BN;
    const int wave = tid >> 5;
    const int lane = tid & 31;
    const int wm   = wave >> 2;   // 0..1 : 64 rows
    const int wn   = wave & 3;    // 0..3 : 32 cols
    const int half = lane >> 4;
    const int l15  = lane & 15;

    // Stage one 128x32 bf16 tile pair: 512 16B chunks each, 2+2 per thread.
    auto stage = [&](int sel, int k0) {
#pragma unroll
        for (int i = 0; i < 2; ++i) {
            int c = tid + i * 256;
            int row = c >> 2, part = c & 3;          // part = 8-bf16 (16B) chunk
            async_copy16(&As[sel][row][part * 8],
                         xh_bf + (size_t)(bm + row) * 2048 + k0 + part * 8);
            async_copy16(&Bs[sel][row][part * 8],
                         wt_bf + (size_t)(bn + row) * 2048 + k0 + part * 8);
        }
    };

    const v8f vzero = {0.f,0.f,0.f,0.f,0.f,0.f,0.f,0.f};
    v8f acc[4][2];
#pragma unroll
    for (int mt = 0; mt < 4; ++mt)
#pragma unroll
        for (int nt = 0; nt < 2; ++nt) acc[mt][nt] = vzero;

    stage(0, 0);
    wait_async_all();
    __syncthreads();

    int cur = 0;
    for (int k0 = 0; k0 < 2048; k0 += BK) {
        if (k0 + BK < 2048) stage(cur ^ 1, k0 + BK);   // prefetch next tile

#pragma unroll
        for (int mt = 0; mt < 4; ++mt) {
            ABFrag a;                                   // A: lane=M, K by half
            int arow = wm * 64 + mt * 16 + l15;
            a.h[0] = *(const v8bf*)&As[cur][arow][8 * half];
            a.h[1] = *(const v8bf*)&As[cur][arow][16 + 8 * half];
#pragma unroll
            for (int nt = 0; nt < 2; ++nt) {
                ABFrag b;                               // B: lane=N, K by half
                int bcol = wn * 32 + nt * 16 + l15;
                b.h[0] = *(const v8bf*)&Bs[cur][bcol][8 * half];
                b.h[1] = *(const v8bf*)&Bs[cur][bcol][16 + 8 * half];
                acc[mt][nt] = __builtin_amdgcn_wmma_f32_16x16x32_bf16(
                    false, a.v, false, b.v, (short)0, acc[mt][nt], false, false);
            }
        }
        wait_async_all();
        __syncthreads();
        cur ^= 1;
    }

    // C/D layout: VGPR j -> M=j (lanes 0-15) / M=j+8 (lanes 16-31), N=lane&15
#pragma unroll
    for (int mt = 0; mt < 4; ++mt)
#pragma unroll
        for (int nt = 0; nt < 2; ++nt) {
            int col = bn + wn * 32 + nt * 16 + l15;
            float bv = bias[col];
#pragma unroll
            for (int j = 0; j < 8; ++j) {
                int row = bm + wm * 64 + mt * 16 + j + 8 * half;
                out[(size_t)row * 3072 + col] = acc[mt][nt][j] + bv;
            }
        }
}

// ---------------------------------------------------------------------------
// Kernel 2: w = sigmoid(xh @ Ww + bw), r = sigmoid(xh @ Wr + br)   (1024 x 16)
// ---------------------------------------------------------------------------
__global__ __launch_bounds__(256) void gates(
    const float* __restrict__ x, const float* __restrict__ h,
    const float* __restrict__ Ww, const float* __restrict__ Wwb,
    const float* __restrict__ Wr, const float* __restrict__ Wrb,
    float* __restrict__ wbuf, float* __restrict__ rbuf)
{
    __shared__ float xh[2048];
    const int b = blockIdx.x, tid = threadIdx.x;
    for (int i = tid; i < 1024; i += 256) {
        xh[i]        = x[(size_t)b * 1024 + i];
        xh[1024 + i] = h[(size_t)b * 1024 + i];
    }
    __syncthreads();

    const int wave = tid >> 5, lane = tid & 31;
    for (int o = wave; o < 32; o += 8) {
        const int head = o & 15;
        const float* Wc = (o < 16) ? Ww : Wr;
        const float  bb = (o < 16) ? Wwb[head] : Wrb[head];
        float s = 0.f;
        for (int i = lane; i < 2048; i += 32)
            s += xh[i] * Wc[(size_t)i * 16 + head];
#pragma unroll
        for (int off = 16; off >= 1; off >>= 1) s += __shfl_xor(s, off, 32);
        if (lane == 0) {
            float val = 1.f / (1.f + __expf(-(s + bb)));
            if (o < 16) wbuf[b * 16 + head] = val;
            else        rbuf[b * 16 + head] = val;
        }
    }
}

// ---------------------------------------------------------------------------
// Kernel 3: per (b, head): normalize q,k; v_r = AM@k; vp = w*(v - v_r);
// AM_new = AM + outer(vp, k); h_new = (AM_new@q) * r.  Single pass over AM —
// 512 MB of HBM traffic, the ~22 us roofline floor for this whole op.
// ---------------------------------------------------------------------------
__global__ __launch_bounds__(256) void am_update(
    const float* __restrict__ AM, const float* __restrict__ qkv,
    const float* __restrict__ wbuf, const float* __restrict__ rbuf,
    float* __restrict__ hout, float* __restrict__ amout)
{
    __shared__ float qs[64], ks[64], vs[64], redq[64], redk[64];
    const int bn = blockIdx.x;            // b*16 + n
    const int b = bn >> 4, n = bn & 15;
    const int tid = threadIdx.x;

    const float* qkvrow = qkv + (size_t)b * 3072;
    if (tid < 64) {
        float qv = qkvrow[n * 64 + tid];
        float kv = qkvrow[1024 + n * 64 + tid];
        vs[tid] = qkvrow[2048 + n * 64 + tid];
        qs[tid] = qv; ks[tid] = kv;
        redq[tid] = qv * qv; redk[tid] = kv * kv;
    }
    __syncthreads();
#pragma unroll
    for (int off = 32; off >= 1; off >>= 1) {
        if (tid < off) { redq[tid] += redq[tid + off]; redk[tid] += redk[tid + off]; }
        __syncthreads();
    }
    if (tid < 64) {
        float rq = 1.f / fmaxf(sqrtf(redq[0]), 1e-12f);
        float rk = 1.f / fmaxf(sqrtf(redk[0]), 1e-12f);
        qs[tid] *= rq; ks[tid] *= rk;
    }
    __syncthreads();

    const float w = wbuf[bn];
    const float r = rbuf[bn];
    const int row = tid >> 2, sub = tid & 3;
    const size_t base = (size_t)bn * 4096 + (size_t)row * 64;

    float4 amv[4];
    float vr = 0.f;
#pragma unroll
    for (int i = 0; i < 4; ++i) {
        int c = i * 16 + sub * 4;                 // fully coalesced float4 loads
        amv[i] = *(const float4*)&AM[base + c];
        vr += amv[i].x * ks[c] + amv[i].y * ks[c + 1]
            + amv[i].z * ks[c + 2] + amv[i].w * ks[c + 3];
    }
    vr += __shfl_xor(vr, 1, 32);
    vr += __shfl_xor(vr, 2, 32);

    const float vp = w * (vs[row] - vr);
    float hn = 0.f;
#pragma unroll
    for (int i = 0; i < 4; ++i) {
        int c = i * 16 + sub * 4;
        amv[i].x += ks[c]     * vp;
        amv[i].y += ks[c + 1] * vp;
        amv[i].z += ks[c + 2] * vp;
        amv[i].w += ks[c + 3] * vp;
        hn += amv[i].x * qs[c] + amv[i].y * qs[c + 1]
            + amv[i].z * qs[c + 2] + amv[i].w * qs[c + 3];
        *(float4*)&amout[base + c] = amv[i];
    }
    hn += __shfl_xor(hn, 1, 32);
    hn += __shfl_xor(hn, 2, 32);
    if (sub == 0)
        hout[(size_t)b * 1024 + n * 64 + row] = hn * r;
}

// ---------------------------------------------------------------------------
extern "C" void kernel_launch(void* const* d_in, const int* in_sizes, int n_in,
                              void* d_out, int out_size, void* d_ws, size_t ws_size,
                              hipStream_t stream)
{
    const float* x      = (const float*)d_in[0];
    const float* h      = (const float*)d_in[1];
    const float* AM     = (const float*)d_in[2];
    const float* Wqkv_w = (const float*)d_in[3];
    const float* Wqkv_b = (const float*)d_in[4];
    const float* Ww_w   = (const float*)d_in[5];
    const float* Ww_b   = (const float*)d_in[6];
    const float* Wr_w   = (const float*)d_in[7];
    const float* Wr_b   = (const float*)d_in[8];

    float* hout  = (float*)d_out;                    // (1024, 1024)
    float* amout = hout + (size_t)1024 * 1024;       // (1024,16,64,64)

    float* qkv  = (float*)d_ws;                      // 1024*3072 f32
    float* wbuf = qkv + (size_t)1024 * 3072;         // 1024*16 f32
    float* rbuf = wbuf + 1024 * 16;                  // 1024*16 f32
    unsigned short* xh_bf = (unsigned short*)(rbuf + 1024 * 16); // (1024,2048) bf16
    unsigned short* wt_bf = xh_bf + (size_t)1024 * 2048;         // (3072,2048) bf16

    pack_xh_bf16<<<4096, 256, 0, stream>>>(x, h, xh_bf);
    transpose_w_bf16<<<dim3(3072 / 32, 2048 / 32), 256, 0, stream>>>(Wqkv_w, wt_bf);
    qkv_gemm<<<dim3(3072 / BN, 1024 / BM), 256, 0, stream>>>(
        xh_bf, wt_bf, Wqkv_b, qkv);
    gates<<<1024, 256, 0, stream>>>(x, h, Ww_w, Ww_b, Wr_w, Wr_b, wbuf, rbuf);
    am_update<<<1024 * 16, 256, 0, stream>>>(AM, qkv, wbuf, rbuf, hout, amout);
}
